// MultiScaleLDSA_Lite_20100446945673
// MI455X (gfx1250) — compile-verified, matrix operand order checked by reading, not run on hardware
//
#include <hip/hip_runtime.h>

// ---------------------------------------------------------------------------
// MultiScaleLDSA_Lite for MI455X (gfx1250, wave32, WMMA + async-to-LDS)
// B=2, T=2048, D=1024, N_HEAD=16, contexts (21,41), heads/scale=8, hd=128
// ---------------------------------------------------------------------------

typedef __attribute__((ext_vector_type(16))) __bf16 v16bf;
typedef __attribute__((ext_vector_type(8)))  __bf16 v8bf;
typedef __attribute__((ext_vector_type(8)))  float  v8f;

#define Bc   2
#define Tc   2048
#define Dc   1024
#define BTc  (Bc * Tc)          // 4096 rows
#define NATT 656                // 16 heads * 41 -> 41 N-tiles of 16

// LDS byte-offset of a __shared__ object (addrspacecast to LDS, then to u32).
__device__ __forceinline__ unsigned lds_off(const void* p) {
    return (unsigned)(unsigned long long)
        (const __attribute__((address_space(3))) void*)p;
}

// ------------------------------- f32 -> bf16 -------------------------------
__global__ void cvt_bf16_kernel(const float* __restrict__ in,
                                __bf16* __restrict__ out, int n4) {
    int i = blockIdx.x * blockDim.x + threadIdx.x;
    if (i >= n4) return;
    const float4 v = ((const float4*)in)[i];
    __bf16* o = out + (size_t)i * 4;
    o[0] = (__bf16)v.x; o[1] = (__bf16)v.y; o[2] = (__bf16)v.z; o[3] = (__bf16)v.w;
}

// --------------------------- WMMA bf16 GEMM --------------------------------
// C[M,N] (f32) = A[M,K] (bf16, row-major) * W[N,K]^T (bf16, row-major rows=N)
// One wave computes a 16 x (16*NSUB) strip: NSUB accumulators share one A
// fragment per K-step (4x fewer A fragment loads for NSUB=4).
template <int NSUB>
__global__ void gemm_bf16_wmma(const __bf16* __restrict__ A,
                               const __bf16* __restrict__ W,
                               float* __restrict__ C,
                               __bf16* __restrict__ Crelu,
                               int M, int N, int K) {
    const int lane   = threadIdx.x & 31;
    const int wid    = blockIdx.x * (blockDim.x >> 5) + (threadIdx.x >> 5);
    const int groups = (N >> 4) / NSUB;       // N-strip groups
    const int nt0    = (wid % groups) * NSUB;
    const int mt     = wid / groups;
    if (mt >= (M >> 4)) return;               // wave-uniform: EXEC stays all-1s

    const int r     = lane & 15;              // A: M-row in tile / B: N-col in tile
    const int khalf = lane >> 4;              // K-half select per ISA 16-bit layout

    const __bf16* __restrict__ arow = A + (size_t)(mt * 16 + r) * K;
    const __bf16* __restrict__ brow = W + (size_t)(nt0 * 16 + r) * K;

    v8f acc[NSUB];
#pragma unroll
    for (int s = 0; s < NSUB; ++s) acc[s] = (v8f){};

    for (int k0 = 0; k0 < K; k0 += 32) {
        v16bf a;
        // elements 0..7  (VGPR0-3): K = k0 + khalf*8 + 0..7
        // elements 8..15 (VGPR4-7): K = k0 + 16 + khalf*8 + 0..7
        *(v8bf*)&a       = *(const v8bf*)(arow + k0 + khalf * 8);
        *((v8bf*)&a + 1) = *(const v8bf*)(arow + k0 + 16 + khalf * 8);
#pragma unroll
        for (int s = 0; s < NSUB; ++s) {
            const __bf16* bs = brow + (size_t)s * 16 * K;
            v16bf b;
            *(v8bf*)&b       = *(const v8bf*)(bs + k0 + khalf * 8);
            *((v8bf*)&b + 1) = *(const v8bf*)(bs + k0 + 16 + khalf * 8);
            acc[s] = __builtin_amdgcn_wmma_f32_16x16x32_bf16(
                /*neg_a=*/false, a, /*neg_b=*/false, b,
                /*c_mod=*/(short)0, acc[s], /*reuse_a=*/false, /*reuse_b=*/false);
        }
    }

    // C/D layout: lane L, VGPR v -> (m = khalf*8 + v, n = L&15)
    const size_t row0 = (size_t)(mt * 16 + khalf * 8);
#pragma unroll
    for (int s = 0; s < NSUB; ++s) {
        const size_t col = (size_t)((nt0 + s) * 16 + r);
        float* crow = C + row0 * N + col;
#pragma unroll
        for (int v = 0; v < 8; ++v) {
            const float val = acc[s][v];
            crow[(size_t)v * N] = val;
            if (Crelu)
                Crelu[(row0 + v) * N + col] = (__bf16)fmaxf(val, 0.0f);
        }
    }
}

// --------------------------- windowed softmax ------------------------------
// attn row layout: [0..167] = scale0 (8 heads x 21), [328..655] = scale1 (8x41).
// Applies softmax over each head's window, scaled by softmax(scale_weights)[i].
__global__ void attn_softmax_kernel(float* __restrict__ attn,
                                    const float* __restrict__ scale_w) {
    int idx = blockIdx.x * blockDim.x + threadIdx.x;
    if (idx >= BTc * 16) return;
    const int row   = idx >> 4;
    const int head  = idx & 15;
    const int scale = head >> 3;
    const int h     = head & 7;
    const int c     = scale ? 41 : 21;
    float* p = attn + (size_t)row * NATT + (scale ? 328 + h * 41 : h * 21);

    const float s0 = scale_w[0], s1 = scale_w[1];
    const float sm = fmaxf(s0, s1);
    const float e0 = __expf(s0 - sm), e1 = __expf(s1 - sm);
    const float sw = (scale ? e1 : e0) / (e0 + e1);

    float mx = -3.0e38f;
    for (int j = 0; j < c; ++j) mx = fmaxf(mx, p[j]);
    float sum = 0.0f;
    for (int j = 0; j < c; ++j) sum += __expf(p[j] - mx);
    const float inv = sw / sum;
    for (int j = 0; j < c; ++j) p[j] = __expf(p[j] - mx) * inv;
}

// ----------------------------- aggregation ---------------------------------
// out[b,t,h*128+j] = sum_scales sum_c w[b,t,h,c] * x[b, t+c-pad, h*128+j]
// One block (256 thr) per (b,t); each thread owns 4 features.
// Weights are staged in LDS via the CDNA5 async-to-LDS path (ASYNCcnt).
__global__ void aggregate_kernel(const float* __restrict__ x,
                                 const float* __restrict__ attn,
                                 __bf16* __restrict__ outb) {
    __shared__ __align__(16) float wsm[496];   // 8*21 + 8*41
    const int bt  = blockIdx.x;                // b*T + t
    const int t   = bt & (Tc - 1);
    const int tid = threadIdx.x;

    const float* __restrict__ arow = attn + (size_t)bt * NATT;
    for (int i = tid; i < 496; i += 256) {
        const float* g = arow + (i < 168 ? i : i + 160);  // skip unused [168,328)
        const unsigned la = lds_off(&wsm[i]);
        asm volatile("global_load_async_to_lds_b32 %0, %1, off"
                     :
                     : "v"(la), "v"(g)
                     : "memory");
    }
    asm volatile("s_wait_asynccnt 0" ::: "memory");
    __syncthreads();

    const int f0 = tid * 4;
    const int h  = tid >> 5;                   // head = (tid*4)/128
    float a0 = 0.f, a1 = 0.f, a2 = 0.f, a3 = 0.f;
    const float* __restrict__ xb = x + (size_t)(bt - t) * Dc;  // batch base

#pragma unroll
    for (int cc = 0; cc < 21; ++cc) {
        const int tt = t + cc - 10;
        if ((unsigned)tt < (unsigned)Tc) {
            const float4 xv = *(const float4*)(xb + (size_t)tt * Dc + f0);
            const float w = wsm[h * 21 + cc];
            a0 += w * xv.x; a1 += w * xv.y; a2 += w * xv.z; a3 += w * xv.w;
        }
    }
#pragma unroll
    for (int cc = 0; cc < 41; ++cc) {
        const int tt = t + cc - 20;
        if ((unsigned)tt < (unsigned)Tc) {
            const float4 xv = *(const float4*)(xb + (size_t)tt * Dc + f0);
            const float w = wsm[168 + h * 41 + cc];
            a0 += w * xv.x; a1 += w * xv.y; a2 += w * xv.z; a3 += w * xv.w;
        }
    }
    __bf16* o = outb + (size_t)bt * Dc + f0;
    o[0] = (__bf16)a0; o[1] = (__bf16)a1; o[2] = (__bf16)a2; o[3] = (__bf16)a3;
}

// ------------------------------- launcher ----------------------------------
extern "C" void kernel_launch(void* const* d_in, const int* in_sizes, int n_in,
                              void* d_out, int out_size, void* d_ws, size_t ws_size,
                              hipStream_t stream) {
    const float* q       = (const float*)d_in[0];   // [2,2048,1024]
    // d_in[1] = key, d_in[2] = value : unused by the reference
    const float* w_in    = (const float*)d_in[3];   // [1024,1024]
    const float* w_attn  = (const float*)d_in[4];   // [656,1024]
    const float* w_out   = (const float*)d_in[5];   // [1024,1024]
    const float* scale_w = (const float*)d_in[6];   // [2]
    float* out           = (float*)d_out;           // [2,2048,1024] f32

    char* ws = (char*)d_ws;
    size_t off = 0;
    auto alloc = [&](size_t bytes) { void* p = ws + off; off = (off + bytes + 1023) & ~(size_t)1023; return p; };

    __bf16* qb   = (__bf16*)alloc((size_t)BTc * Dc * 2);   // query bf16
    __bf16* wib  = (__bf16*)alloc((size_t)Dc * Dc * 2);    // W_in bf16
    __bf16* wab  = (__bf16*)alloc((size_t)NATT * Dc * 2);  // W_attn bf16
    __bf16* wob  = (__bf16*)alloc((size_t)Dc * Dc * 2);    // W_out bf16
    float*  xf   = (float*)alloc((size_t)BTc * Dc * 4);    // x f32
    __bf16* rxb  = (__bf16*)alloc((size_t)BTc * Dc * 2);   // relu(x) bf16
    float*  attn = (float*)alloc((size_t)BTc * NATT * 4);  // attn f32
    __bf16* outb = (__bf16*)alloc((size_t)BTc * Dc * 2);   // aggregated bf16
    (void)ws_size; (void)in_sizes; (void)n_in; (void)out_size;

    // 1) convert operands to bf16
    {
        int n4;
        n4 = (BTc * Dc) / 4;
        cvt_bf16_kernel<<<(n4 + 255) / 256, 256, 0, stream>>>(q, qb, n4);
        n4 = (Dc * Dc) / 4;
        cvt_bf16_kernel<<<(n4 + 255) / 256, 256, 0, stream>>>(w_in, wib, n4);
        n4 = (NATT * Dc) / 4;
        cvt_bf16_kernel<<<(n4 + 255) / 256, 256, 0, stream>>>(w_attn, wab, n4);
        n4 = (Dc * Dc) / 4;
        cvt_bf16_kernel<<<(n4 + 255) / 256, 256, 0, stream>>>(w_out, wob, n4);
    }

    // 2) x = q . W_in^T  (also emit relu(x) in bf16)   [16x64 strips per wave]
    {
        int waves = (BTc / 16) * ((Dc / 16) / 4);
        gemm_bf16_wmma<4><<<(waves + 7) / 8, 256, 0, stream>>>(qb, wib, xf, rxb,
                                                               BTc, Dc, Dc);
    }

    // 3) attn = relu(x) . W_attn^T   (41 N-tiles -> 16x16 per wave)
    {
        int waves = (BTc / 16) * (NATT / 16);
        gemm_bf16_wmma<1><<<(waves + 7) / 8, 256, 0, stream>>>(rxb, wab, attn, nullptr,
                                                               BTc, NATT, Dc);
    }

    // 4) per-(t,head) windowed softmax scaled by softmax(scale_weights)
    attn_softmax_kernel<<<(BTc * 16 + 255) / 256, 256, 0, stream>>>(attn, scale_w);

    // 5) multi-scale windowed aggregation -> bf16 (async-to-LDS weight staging)
    aggregate_kernel<<<BTc, 256, 0, stream>>>(xf, attn, outb);

    // 6) result = out . W_out^T  (f32 to d_out)   [16x64 strips per wave]
    {
        int waves = (BTc / 16) * ((Dc / 16) / 4);
        gemm_bf16_wmma<4><<<(waves + 7) / 8, 256, 0, stream>>>(outb, wob, out,
                                                               nullptr, BTc, Dc, Dc);
    }
}
